// MultiHeadAttention_61658550501423
// MI455X (gfx1250) — matrix operand order repaired and checked by reading, not compiled, not run on hardware
//
#include <hip/hip_runtime.h>
#include <hip/hip_bf16.h>
#include <cstdint>

// ---------------- problem constants ----------------
constexpr int Bb = 2;        // batch
constexpr int Ss = 2048;     // sequence
constexpr int Hh = 16;       // heads
constexpr int Dh = 64;       // head dim
constexpr int DM = 1024;     // model dim

typedef __attribute__((ext_vector_type(16))) __bf16        v16bf;
typedef __attribute__((ext_vector_type(8)))  float         v8f;
typedef __attribute__((ext_vector_type(4)))  unsigned int  u32x4;

union Frag {               // 16 bf16 halves = 8 VGPRs per lane
    v16bf          v;
    u32x4          u[2];
    unsigned short s[16];
};

union Frag8 {              // 8 bf16 halves = 16B
    u32x4          u;
    unsigned short s[8];
};

static __device__ __forceinline__ unsigned short f2bf(float f) {
    union { float f; unsigned u; } a; a.f = f;
    unsigned r = a.u + 0x7fffu + ((a.u >> 16) & 1u);   // round-to-nearest-even
    return (unsigned short)(r >> 16);
}

static __device__ __forceinline__ v8f wmma_bf(v16bf a, v16bf b, v8f c) {
    // D = A(16x32 bf16) x B(32x16 bf16) + C(16x16 f32)
    return __builtin_amdgcn_wmma_f32_16x16x32_bf16(false, a, false, b,
                                                   (short)0, c, false, false);
}

// ---------------- stage 0: weight f32 -> bf16 ----------------
__global__ void cvt_f32_bf16(const float* __restrict__ src,
                             unsigned short* __restrict__ dst, int n) {
    int i = blockIdx.x * blockDim.x + threadIdx.x;
    int stride = gridDim.x * blockDim.x;
    for (; i < n; i += stride) dst[i] = f2bf(src[i]);
}

// ---------------- stage 1/3: GEMM  Y = X @ W^T + b ----------------
// Each wave computes a 16(M) x 64(N) strip: one A fragment feeds 4 WMMAs.
// A_F32: activation source is fp32 (convert on the fly) else bf16.
// SPLIT: write bf16 in (B,H,S,64) split-head layout, else fp32 row-major.
template <bool A_F32, bool SPLIT>
__global__ void __launch_bounds__(256)
proj_gemm(const void* __restrict__ Ap, const unsigned short* __restrict__ W,
          const float* __restrict__ bias, void* __restrict__ Out, int M) {
    const int lane = threadIdx.x & 31;
    const int wid  = threadIdx.x >> 5;
    const int lm   = lane & 15;
    const int hi   = lane >> 4;
    const int ntn  = DM / 64;                       // 64-wide N strips

    int gw = blockIdx.x * (blockDim.x >> 5) + wid;
    int total = (M / 16) * ntn;
    if (gw >= total) return;
    int tm = gw / ntn, tn = gw % ntn;

    const int arow = tm * 16 + lm;                  // A-fragment: lane = row
    const unsigned short* wr[4];
    #pragma unroll
    for (int t = 0; t < 4; t++)                     // B-fragment: lane = out-feature
        wr[t] = W + (size_t)(tn * 64 + t * 16 + lm) * DM;

    v8f acc[4] = {};
    for (int k0 = 0; k0 < DM; k0 += 32) {
        Frag a;
        if constexpr (A_F32) {
            const float* ar = (const float*)Ap + (size_t)arow * DM;
            #pragma unroll
            for (int i = 0; i < 8; i++) {
                a.s[i]     = f2bf(ar[k0 + hi * 8 + i]);        // K = hi*8 + h
                a.s[8 + i] = f2bf(ar[k0 + 16 + hi * 8 + i]);   // K = 16 + hi*8 + h
            }
        } else {
            const unsigned short* ar = (const unsigned short*)Ap + (size_t)arow * DM;
            a.u[0] = *(const u32x4*)(ar + k0 + hi * 8);
            a.u[1] = *(const u32x4*)(ar + k0 + 16 + hi * 8);
        }
        #pragma unroll
        for (int t = 0; t < 4; t++) {
            Frag b;                                  // B halves: K = hi*8 + h, 16 + hi*8 + h
            b.u[0] = *(const u32x4*)(wr[t] + k0 + hi * 8);
            b.u[1] = *(const u32x4*)(wr[t] + k0 + 16 + hi * 8);
            acc[t] = wmma_bf(a.v, b.v, acc[t]);
        }
    }

    #pragma unroll
    for (int t = 0; t < 4; t++) {
        #pragma unroll
        for (int v = 0; v < 8; v++) {               // D row = v + 8*hi, col = lm
            int m = tm * 16 + v + 8 * hi;
            int n = tn * 64 + t * 16 + lm;
            float val = acc[t][v] + bias[n];
            if constexpr (SPLIT) {
                int bb = m / Ss, s = m % Ss;
                int h = n >> 6, d = n & 63;
                ((unsigned short*)Out)[(((size_t)(bb * Hh + h)) * Ss + s) * Dh + d] = f2bf(val);
            } else {
                ((float*)Out)[(size_t)m * DM + n] = val;
            }
        }
    }
}

// ---------------- stage 2: flash attention ----------------
// Block = 8 waves sharing one (b,h) and a 128-query span. Per 32-key tile:
// K staged to LDS with async-to-LDS copies, V staged pre-transposed so its
// B fragments are contiguous ds_load_b128. Online softmax per 16-query strip.
__global__ void __launch_bounds__(256)
attn_kernel(const unsigned short* __restrict__ Q, const unsigned short* __restrict__ K,
            const unsigned short* __restrict__ V, const int* __restrict__ valid_len,
            unsigned short* __restrict__ Oa) {
    // LDS: Kt 32x64 bf16 (4KB) | Vt 64x32 bf16 transposed (4KB) | P 8x16x32 (8KB)
    __shared__ unsigned short smem[8192];
    unsigned short* Kt = smem;                       // [key][hd]    32*64
    unsigned short* Vt = smem + 2048;                // [hd][key]    64*32
    unsigned short* Pb = smem + 4096;                // per-wave 16*32

    const int tid  = threadIdx.x;
    const int lane = tid & 31;
    const int wid  = tid >> 5;
    const int lm   = lane & 15;
    const int hi   = lane >> 4;

    const int qchunks = Ss / 128;                    // 16
    int bh = blockIdx.x / qchunks;
    int qbase = (blockIdx.x % qchunks) * 128;
    int b = bh >> 4, h = bh & 15;
    int q0 = qbase + wid * 16;

    int valid = valid_len[b];
    int kend      = min(q0 + 16, valid);             // this wave's key bound
    int kend_blk  = min(qbase + 128, valid);         // block's key bound
    int nkt       = (kend_blk + 31) / 32;            // shared 32-key tiles

    // Q A-fragments (reused for every key tile): lane = query row q0+lm
    const unsigned short* qr = Q + ((size_t)bh * Ss + (q0 + lm)) * Dh;
    Frag qa0, qa1;
    qa0.u[0] = *(const u32x4*)(qr + hi * 8);
    qa0.u[1] = *(const u32x4*)(qr + 16 + hi * 8);
    qa1.u[0] = *(const u32x4*)(qr + 32 + hi * 8);
    qa1.u[1] = *(const u32x4*)(qr + 48 + hi * 8);

    float mrun[8], lrun[8];
    #pragma unroll
    for (int v = 0; v < 8; v++) { mrun[v] = -3.0e38f; lrun[v] = 0.0f; }
    v8f o[4] = {};                                   // O tile 16x64
    unsigned short* pb = Pb + wid * (16 * 32);

    // staging coordinates: thread -> (key row kk, 8-element segment seg)
    const int kk  = tid >> 3;                        // 0..31
    const int seg = tid & 7;                         // 0..7

    for (int kt = 0; kt < nkt; kt++) {
        int k0 = kt * 32;
        __syncthreads();                             // LDS free for overwrite

        // ---- K tile: async global->LDS copy, 16B per thread (ASYNCcnt)
        {
            unsigned lds_off = (unsigned)((kk * 64 + seg * 8) * 2);   // Kt at LDS 0
            const unsigned short* gk =
                K + ((size_t)bh * Ss + k0 + kk) * Dh + seg * 8;
            asm volatile("global_load_async_to_lds_b128 %0, %1, off"
                         :: "v"(lds_off), "v"(gk) : "memory");
        }
        // ---- V tile: load 16B and scatter transposed (Vt[d][k])
        {
            Frag8 r;
            r.u = *(const u32x4*)(V + ((size_t)bh * Ss + k0 + kk) * Dh + seg * 8);
            #pragma unroll
            for (int i = 0; i < 8; i++)
                Vt[(seg * 8 + i) * 32 + kk] = r.s[i];
        }
        asm volatile("s_wait_asynccnt 0x0" ::: "memory");  // own async done
        __syncthreads();                                   // tile visible to all

        if (k0 < kend) {
            // ---- scores: two 16x16 tiles over head-dim (2 wmma steps each)
            v8f sc[2] = {};
            #pragma unroll
            for (int c = 0; c < 2; c++) {
                const unsigned short* kr = Kt + (c * 16 + lm) * Dh;
                Frag kb0, kb1;                       // B halves: K = hi*8 + h, 16 + hi*8 + h
                kb0.u[0] = *(const u32x4*)(kr + hi * 8);
                kb0.u[1] = *(const u32x4*)(kr + 16 + hi * 8);
                kb1.u[0] = *(const u32x4*)(kr + 32 + hi * 8);
                kb1.u[1] = *(const u32x4*)(kr + 48 + hi * 8);
                sc[c] = wmma_bf(qa0.v, kb0.v, sc[c]);
                sc[c] = wmma_bf(qa1.v, kb1.v, sc[c]);
            }

            // ---- mask + online softmax (row = v + 8*hi in 16-lane groups)
            #pragma unroll
            for (int v = 0; v < 8; v++) {
                int qrow = q0 + v + 8 * hi;
                float x0 = sc[0][v] * 0.125f;        // 1/sqrt(64)
                float x1 = sc[1][v] * 0.125f;
                int kc0 = k0 + lm, kc1 = k0 + 16 + lm;
                if (kc0 > qrow || kc0 >= valid) x0 = -3.0e38f;
                if (kc1 > qrow || kc1 >= valid) x1 = -3.0e38f;

                float rmax = fmaxf(x0, x1);
                rmax = fmaxf(rmax, __shfl_xor(rmax, 1, 32));
                rmax = fmaxf(rmax, __shfl_xor(rmax, 2, 32));
                rmax = fmaxf(rmax, __shfl_xor(rmax, 4, 32));
                rmax = fmaxf(rmax, __shfl_xor(rmax, 8, 32));

                float mnew  = fmaxf(mrun[v], rmax);
                float alpha = __expf(mrun[v] - mnew);
                float p0 = __expf(x0 - mnew);
                float p1 = __expf(x1 - mnew);
                float rsum = p0 + p1;
                rsum += __shfl_xor(rsum, 1, 32);
                rsum += __shfl_xor(rsum, 2, 32);
                rsum += __shfl_xor(rsum, 4, 32);
                rsum += __shfl_xor(rsum, 8, 32);
                lrun[v] = lrun[v] * alpha + rsum;
                mrun[v] = mnew;
                #pragma unroll
                for (int t = 0; t < 4; t++) o[t][v] *= alpha;

                int ml = v + 8 * hi;
                pb[ml * 32 + lm]      = f2bf(p0);
                pb[ml * 32 + 16 + lm] = f2bf(p1);
            }

            // wave-private P relayout: D-layout -> A-layout via LDS
            asm volatile("s_wait_dscnt 0x0" ::: "memory");
            Frag pf;
            pf.u[0] = *(const u32x4*)(pb + lm * 32 + hi * 8);
            pf.u[1] = *(const u32x4*)(pb + lm * 32 + 16 + hi * 8);

            // ---- O += P(16x32) @ V(32x64): contiguous frags from Vt
            #pragma unroll
            for (int t = 0; t < 4; t++) {
                const unsigned short* vr = Vt + (t * 16 + lm) * 32;
                Frag vf;                             // B halves: K = hi*8 + h, 16 + hi*8 + h
                vf.u[0] = *(const u32x4*)(vr + hi * 8);
                vf.u[1] = *(const u32x4*)(vr + 16 + hi * 8);
                o[t] = wmma_bf(pf.v, vf.v, o[t]);
            }
        }
    }

    // ---- epilogue: normalize and store merged-head bf16 (B,S,1024)
    #pragma unroll
    for (int v = 0; v < 8; v++) {
        float inv = 1.0f / lrun[v];
        int m = q0 + v + 8 * hi;
        size_t base = ((size_t)b * Ss + m) * DM + h * Dh;
        #pragma unroll
        for (int t = 0; t < 4; t++)
            Oa[base + t * 16 + lm] = f2bf(o[t][v] * inv);
    }
}

// ---------------- host-side launch ----------------
extern "C" void kernel_launch(void* const* d_in, const int* in_sizes, int n_in,
                              void* d_out, int out_size, void* d_ws, size_t ws_size,
                              hipStream_t stream) {
    const float* Xq = (const float*)d_in[0];
    const float* Xk = (const float*)d_in[1];
    const float* Xv = (const float*)d_in[2];
    const int*   vl = (const int*)  d_in[3];
    const float* wq = (const float*)d_in[4];
    const float* bq = (const float*)d_in[5];
    const float* wk = (const float*)d_in[6];
    const float* bk = (const float*)d_in[7];
    const float* wv = (const float*)d_in[8];
    const float* bv = (const float*)d_in[9];
    const float* wo = (const float*)d_in[10];
    const float* bo = (const float*)d_in[11];
    float* out = (float*)d_out;

    uint8_t* ws = (uint8_t*)d_ws;
    const size_t WSZ = (size_t)DM * DM * 2;                 // 2 MB per weight
    const size_t QSZ = (size_t)Bb * Hh * Ss * Dh * 2;       // 8 MB per tensor
    unsigned short* Wq16 = (unsigned short*)(ws + 0 * WSZ);
    unsigned short* Wk16 = (unsigned short*)(ws + 1 * WSZ);
    unsigned short* Wv16 = (unsigned short*)(ws + 2 * WSZ);
    unsigned short* Wo16 = (unsigned short*)(ws + 3 * WSZ);
    unsigned short* Q16  = (unsigned short*)(ws + 4 * WSZ);
    unsigned short* K16  = (unsigned short*)(ws + 4 * WSZ + 1 * QSZ);
    unsigned short* V16  = (unsigned short*)(ws + 4 * WSZ + 2 * QSZ);
    unsigned short* A16  = (unsigned short*)(ws + 4 * WSZ + 3 * QSZ);

    const int nW = DM * DM;
    cvt_f32_bf16<<<512, 256, 0, stream>>>(wq, Wq16, nW);
    cvt_f32_bf16<<<512, 256, 0, stream>>>(wk, Wk16, nW);
    cvt_f32_bf16<<<512, 256, 0, stream>>>(wv, Wv16, nW);
    cvt_f32_bf16<<<512, 256, 0, stream>>>(wo, Wo16, nW);

    const int M = Bb * Ss;                                   // 4096 rows
    const int gemm_blocks = (M / 16) * (DM / 64) / 8;        // 512
    proj_gemm<true, true ><<<gemm_blocks, 256, 0, stream>>>(Xq, Wq16, bq, Q16, M);
    proj_gemm<true, true ><<<gemm_blocks, 256, 0, stream>>>(Xk, Wk16, bk, K16, M);
    proj_gemm<true, true ><<<gemm_blocks, 256, 0, stream>>>(Xv, Wv16, bv, V16, M);

    const int attn_blocks = Bb * Hh * (Ss / 128);            // 512
    attn_kernel<<<attn_blocks, 256, 0, stream>>>(Q16, K16, V16, vl, A16);

    proj_gemm<false, false><<<gemm_blocks, 256, 0, stream>>>(A16, Wo16, bo, out, M);
}
